// MinGRU_72267119723114
// MI455X (gfx1250) — matrix operand order repaired and checked by reading, not compiled, not run on hardware
//
#include <hip/hip_runtime.h>
#include <hip/hip_bf16.h>
#include <math.h>

// ---------- types for WMMA ----------
typedef __attribute__((ext_vector_type(16))) __bf16 v16bf;
typedef __attribute__((ext_vector_type(8)))  float  v8f;

union FragU {
    uint4 u[2];   // two b128 chunks
    v16bf v;      // 16 bf16 = 32 bytes = 8 VGPRs
};

// ---------- device helpers ----------
__device__ __forceinline__ float softplus_f(float x) {
    // log(1 + e^x), stable
    float ax = fabsf(x);
    return fmaxf(x, 0.0f) + __logf(1.0f + __expf(-ax));
}

__device__ __forceinline__ float lae(float a, float b) {
    // logaddexp; tolerates -inf in one argument
    float m = fmaxf(a, b);
    return m + __logf(__expf(a - m) + __expf(b - m));
}

__device__ __forceinline__ unsigned short f2bf(float f) {
    union { float f; unsigned u; } c; c.f = f;
    unsigned u = c.u;
    unsigned r = (u + 0x7FFFu + ((u >> 16) & 1u)) >> 16;  // RNE
    return (unsigned short)r;
}

// ---------- fp32 -> bf16 conversion (4 elems/thread) ----------
__global__ __launch_bounds__(256)
void cvt_bf16_kernel(const float* __restrict__ src, unsigned short* __restrict__ dst, int n4) {
    int i = blockIdx.x * 256 + threadIdx.x;
    if (i >= n4) return;
    float4 f = reinterpret_cast<const float4*>(src)[i];
    ushort4 o;
    o.x = f2bf(f.x); o.y = f2bf(f.y); o.z = f2bf(f.z); o.w = f2bf(f.w);
    reinterpret_cast<ushort4*>(dst)[i] = o;
}

// ---------- GEMM (bf16 WMMA) + MinGRU gate activation ----------
// A: (16384 x 1024) bf16 row-major. W: (2048 x 1024) bf16 row-major.
// Block: 256 thr = 8 waves as 4(M) x 2(N). Block tile: 128(M) x 64(Ngate).
// Each wave: 2x2 tiles of 16x16 for BOTH gate (col n) and hidden (col n+1024).
// Epilogue writes vc[row*1024+col] = {log_v, log_c} (float2).
__global__ __launch_bounds__(256)
void gemm_act_kernel(const unsigned short* __restrict__ A,
                     const unsigned short* __restrict__ W,
                     const float* __restrict__ bias,
                     float2* __restrict__ vc) {
    const int lane  = threadIdx.x & 31;
    const int wave  = threadIdx.x >> 5;
    const int waveM = wave & 3;
    const int waveN = wave >> 2;
    const int lr = lane & 15;   // row/col-in-tile selector
    const int lh = lane >> 4;   // half selector (0/1)

    const int Mbase = blockIdx.x * 128 + waveM * 32;
    const int Ngate = blockIdx.y * 64 + waveN * 32;

    // A fragment base pointers (per lane): row = Mbase+tm*16+lr, K offset lh*16
    const unsigned short* aptr[2];
    aptr[0] = A + (size_t)(Mbase + 0  + lr) * 1024 + lh * 16;
    aptr[1] = A + (size_t)(Mbase + 16 + lr) * 1024 + lh * 16;
    // B fragment base pointers: col = Ngate+tn*16+lr (gate) / +1024 (hidden), K offset lh*16
    const unsigned short* bgptr[2];
    const unsigned short* bhptr[2];
    bgptr[0] = W + (size_t)(Ngate + 0  + lr) * 1024 + lh * 16;
    bgptr[1] = W + (size_t)(Ngate + 16 + lr) * 1024 + lh * 16;
    bhptr[0] = bgptr[0] + (size_t)1024 * 1024;
    bhptr[1] = bgptr[1] + (size_t)1024 * 1024;

    v8f accg[2][2] = {};
    v8f acch[2][2] = {};

    for (int k = 0; k < 1024; k += 32) {
        FragU a[2], bg[2], bh[2];
#pragma unroll
        for (int t = 0; t < 2; ++t) {
            a[t].u[0]  = *reinterpret_cast<const uint4*>(aptr[t]  + k);
            a[t].u[1]  = *reinterpret_cast<const uint4*>(aptr[t]  + k + 8);
            bg[t].u[0] = *reinterpret_cast<const uint4*>(bgptr[t] + k);
            bg[t].u[1] = *reinterpret_cast<const uint4*>(bgptr[t] + k + 8);
            bh[t].u[0] = *reinterpret_cast<const uint4*>(bhptr[t] + k);
            bh[t].u[1] = *reinterpret_cast<const uint4*>(bhptr[t] + k + 8);
        }
#pragma unroll
        for (int tm = 0; tm < 2; ++tm) {
#pragma unroll
            for (int tn = 0; tn < 2; ++tn) {
                accg[tm][tn] = __builtin_amdgcn_wmma_f32_16x16x32_bf16(
                    false, a[tm].v, false, bg[tn].v, (short)0, accg[tm][tn], false, false);
                acch[tm][tn] = __builtin_amdgcn_wmma_f32_16x16x32_bf16(
                    false, a[tm].v, false, bh[tn].v, (short)0, acch[tm][tn], false, false);
            }
        }
    }

    // Epilogue: bias add + log-space MinGRU gates, fused store of (log_v, log_c).
#pragma unroll
    for (int tn = 0; tn < 2; ++tn) {
        const int col = Ngate + tn * 16 + lr;
        const float bg_b = bias[col];
        const float bh_b = bias[col + 1024];
#pragma unroll
        for (int tm = 0; tm < 2; ++tm) {
            const int row0 = Mbase + tm * 16 + lh * 8;
#pragma unroll
            for (int i = 0; i < 8; ++i) {
                float g  = accg[tm][tn][i] + bg_b;
                float hd = acch[tm][tn][i] + bh_b;
                float log_z = -softplus_f(-g);
                float log_c = -softplus_f(g);
                float lg = (hd >= 0.0f) ? __logf(hd + 0.5f) : -softplus_f(-hd);
                vc[(size_t)(row0 + i) * 1024 + col] = make_float2(log_z + lg, log_c);
            }
        }
    }
}

// ---------- scan phase A: per-chunk composed transform ----------
// 8192 channels x 32 chunks of 64 steps. cv[j*8192 + ch] = (V, C).
__global__ __launch_bounds__(256)
void scanA_kernel(const float2* __restrict__ vc, float2* __restrict__ cv) {
    int tid = blockIdx.x * 256 + threadIdx.x;     // 0..262143
    int h = tid & 1023;
    int b = (tid >> 10) & 7;
    int j = tid >> 13;                            // chunk 0..31
    float C = 0.0f;
    float V = -INFINITY;
    size_t base = (size_t)(b * 2048 + j * 64) * 1024 + h;
    for (int s = 0; s < 64; ++s) {
        float2 p = vc[base + (size_t)s * 1024];
        C += p.y;
        V = lae(V + p.y, p.x);
    }
    cv[(size_t)j * 8192 + b * 1024 + h] = make_float2(V, C);
}

// ---------- scan phase B: carry scan over 32 chunks ----------
// rp[j*8192 + ch] = log-state entering chunk j. h0: (B,1,H) -> b*1024+h.
__global__ __launch_bounds__(256)
void scanB_kernel(const float2* __restrict__ cv, const float* __restrict__ h0,
                  float* __restrict__ rp) {
    int tid = blockIdx.x * 256 + threadIdx.x;     // 0..8191
    float r = __logf(h0[tid]);
    for (int j = 0; j < 32; ++j) {
        rp[(size_t)j * 8192 + tid] = r;
        float2 t = cv[(size_t)j * 8192 + tid];
        r = lae(r + t.y, t.x);
    }
}

// ---------- scan phase C: apply within chunk, exp + residual ----------
// dst[row] = exp(r) + addsrc[row]; hfin[b*1024+h] = exp(r) at s = S-1.
__global__ __launch_bounds__(256)
void scanC_kernel(const float2* __restrict__ vc, const float* __restrict__ rp,
                  const float* __restrict__ addsrc, float* __restrict__ dst,
                  float* __restrict__ hfin) {
    int tid = blockIdx.x * 256 + threadIdx.x;
    int h = tid & 1023;
    int b = (tid >> 10) & 7;
    int j = tid >> 13;
    float r = rp[(size_t)j * 8192 + b * 1024 + h];
    size_t base = (size_t)(b * 2048 + j * 64) * 1024 + h;
    float e = 0.0f;
    for (int s = 0; s < 64; ++s) {
        size_t idx = base + (size_t)s * 1024;
        float2 p = vc[idx];
        r = lae(r + p.y, p.x);
        e = __expf(r);
        dst[idx] = e + addsrc[idx];
    }
    if (j == 31) hfin[b * 1024 + h] = e;
}

// ---------- host launcher ----------
extern "C" void kernel_launch(void* const* d_in, const int* in_sizes, int n_in,
                              void* d_out, int out_size, void* d_ws, size_t ws_size,
                              hipStream_t stream) {
    (void)in_sizes; (void)n_in; (void)out_size; (void)ws_size;

    const float* x  = (const float*)d_in[0];   // (8,2048,1024)
    const float* hI = (const float*)d_in[1];   // (2,8,1,1024)
    const float* w0 = (const float*)d_in[2];   // (2048,1024)
    const float* b0 = (const float*)d_in[3];   // (2048)
    const float* w1 = (const float*)d_in[4];   // (2048,1024)
    const float* b1 = (const float*)d_in[5];   // (2048)

    float* out   = (float*)d_out;                        // (8,2048,1024)
    float* hfin0 = (float*)d_out + 16777216;             // (8,1024)
    float* hfin1 = (float*)d_out + 16777216 + 8192;      // (8,1024)

    char* ws = (char*)d_ws;
    unsigned short* xb  = (unsigned short*)(ws);                         // 32 MB (x / inp bf16)
    unsigned short* w0b = (unsigned short*)(ws + 33554432);              //  4 MB
    unsigned short* w1b = (unsigned short*)(ws + 37748736);              //  4 MB
    float2*         vc  = (float2*)(ws + 41943040);                      // 128 MB
    float*          inp = (float*)(ws + 176160768);                      //  64 MB
    float2*         cv  = (float2*)(ws + 243269632);                     //   2 MB
    float*          rp  = (float*)(ws + 245366784);                      //   1 MB

    dim3 blk(256);
    dim3 gGemm(128, 16);   // M/128, Ngate/64

    // layer 0
    cvt_bf16_kernel<<<16384, blk, 0, stream>>>(x,  xb,  16777216 / 4);
    cvt_bf16_kernel<<<2048,  blk, 0, stream>>>(w0, w0b, 2097152 / 4);
    cvt_bf16_kernel<<<2048,  blk, 0, stream>>>(w1, w1b, 2097152 / 4);
    gemm_act_kernel<<<gGemm, blk, 0, stream>>>(xb, w0b, b0, vc);
    scanA_kernel<<<1024, blk, 0, stream>>>(vc, cv);
    scanB_kernel<<<32,   blk, 0, stream>>>(cv, hI, rp);
    scanC_kernel<<<1024, blk, 0, stream>>>(vc, rp, x, inp, hfin0);   // inp = out0 + x

    // layer 1 (reuse xb for inp bf16, vc/cv/rp reused)
    cvt_bf16_kernel<<<16384, blk, 0, stream>>>(inp, xb, 16777216 / 4);
    gemm_act_kernel<<<gGemm, blk, 0, stream>>>(xb, w1b, b1, vc);
    scanA_kernel<<<1024, blk, 0, stream>>>(vc, cv);
    scanB_kernel<<<32,   blk, 0, stream>>>(cv, hI + 8192, rp);
    scanC_kernel<<<1024, blk, 0, stream>>>(vc, rp, inp, out, hfin1); // out = out1 + inp
}
